// PointNetPlusPlus_83820581749272
// MI455X (gfx1250) — matrix operand order repaired and checked by reading, not compiled, not run on hardware
//
#include <hip/hip_runtime.h>

typedef __attribute__((ext_vector_type(16))) _Float16 v16h;
typedef __attribute__((ext_vector_type(8)))  float    v8f;

#define EPSV 1e-5f

// ---------------------------------------------------------------------------
// Prep: fold BN into per-channel affine, transpose weight to column-major f16
// Wcm[n*kpad + k] = (k < cin) ? W[k*cout + n] : 0
// ---------------------------------------------------------------------------
__global__ void prep_layer_kernel(const float* __restrict__ W, const float* __restrict__ bias,
                                  const float* __restrict__ g, const float* __restrict__ b,
                                  const float* __restrict__ m, const float* __restrict__ v,
                                  int cin, int cout, int kpad,
                                  _Float16* __restrict__ wcm,
                                  float* __restrict__ scale, float* __restrict__ shift) {
    int tid = blockIdx.x * blockDim.x + threadIdx.x;
    int total = cout * kpad;
    if (tid < total) {
        int n = tid / kpad, k = tid % kpad;
        float w = (k < cin) ? W[k * cout + n] : 0.0f;
        wcm[(size_t)n * kpad + k] = (_Float16)w;
    }
    if (tid < cout) {
        float s = g[tid] * rsqrtf(v[tid] + EPSV);
        scale[tid] = s;
        shift[tid] = (bias[tid] - m[tid]) * s + b[tid];
    }
}

// ---------------------------------------------------------------------------
// Farthest point sampling: single workgroup, dist array in LDS (dynamic).
// Emits idx[it] = far BEFORE the update (matches the lax.scan carry order).
// 65536*4B = 256KB of LDS fits in CDNA5's 320KB/WGP.
// ---------------------------------------------------------------------------
__global__ void fps_kernel(const float* __restrict__ xy, int n, int npoint,
                           int* __restrict__ idx_out) {
    extern __shared__ float dist[];
    __shared__ float redv[1024];
    __shared__ int   redi[1024];
    __shared__ int   s_far;
    int tid = threadIdx.x, nt = blockDim.x;
    for (int i = tid; i < n; i += nt) dist[i] = 1e10f;
    if (tid == 0) s_far = 0;
    __syncthreads();
    for (int it = 0; it < npoint; ++it) {
        int far = s_far;
        if (tid == 0) idx_out[it] = far;
        float fx = xy[2 * far], fy = xy[2 * far + 1];
        float bm = -1.0f; int bi = 0;
        for (int i = tid; i < n; i += nt) {
            float dx = xy[2 * i] - fx, dy = xy[2 * i + 1] - fy;
            float nd = fminf(dist[i], dx * dx + dy * dy);
            dist[i] = nd;
            if (nd > bm) { bm = nd; bi = i; }
        }
        redv[tid] = bm; redi[tid] = bi;
        __syncthreads();
        for (int s = nt >> 1; s > 0; s >>= 1) {
            if (tid < s) {
                float ov = redv[tid + s]; int oi = redi[tid + s];
                if (ov > redv[tid] || (ov == redv[tid] && oi < redi[tid])) {
                    redv[tid] = ov; redi[tid] = oi;
                }
            }
            __syncthreads();
        }
        if (tid == 0) s_far = redi[0];
        __syncthreads();
    }
}

__global__ void gather_xy_kernel(const float* __restrict__ xy, const int* __restrict__ idx,
                                 int n, float* __restrict__ out) {
    int i = blockIdx.x * blockDim.x + threadIdx.x;
    if (i < n) { int p = idx[i]; out[2 * i] = xy[2 * p]; out[2 * i + 1] = xy[2 * p + 1]; }
}

// First `ns` indices (in index order) with sq-dist <= r2; pad with first hit.
__global__ void group_kernel(const float* __restrict__ cxy, int nc,
                             const float* __restrict__ pxy, int np,
                             float r2, int ns, int* __restrict__ gidx) {
    int c = blockIdx.x * blockDim.x + threadIdx.x;
    if (c >= nc) return;
    float cx = cxy[2 * c], cy = cxy[2 * c + 1];
    int cnt = 0;
    for (int i = 0; i < np && cnt < ns; ++i) {
        float dx = pxy[2 * i] - cx, dy = pxy[2 * i + 1] - cy;
        if (dx * dx + dy * dy <= r2) gidx[c * ns + cnt++] = i;
    }
    if (cnt == 0) { gidx[c * ns] = 0; cnt = 1; }
    int first = gidx[c * ns];
    for (int j = cnt; j < ns; ++j) gidx[c * ns + j] = first;
}

// Build SA1 input rows: [dx, dy, 0(z), zeros...] padded to 32, f16.
__global__ void build_a1_kernel(const float* __restrict__ pxy, const float* __restrict__ cxy,
                                const int* __restrict__ gidx, int ncenter, int ns,
                                _Float16* __restrict__ A) {
    int r = blockIdx.x * blockDim.x + threadIdx.x;
    if (r >= ncenter * ns) return;
    int c = r / ns, p = gidx[r];
    _Float16* row = A + (size_t)r * 32;
    row[0] = (_Float16)(pxy[2 * p] - cxy[2 * c]);
    row[1] = (_Float16)(pxy[2 * p + 1] - cxy[2 * c + 1]);
    for (int k = 2; k < 32; ++k) row[k] = (_Float16)0.0f;
}

// Build SA2 input rows: [dx, dy, 0, feat(fdim), zeros] padded to kpad.
__global__ void build_a2_kernel(const float* __restrict__ pxy, const float* __restrict__ cxy,
                                const int* __restrict__ gidx, const _Float16* __restrict__ feat,
                                int fdim, int ncenter, int ns, int kpad,
                                _Float16* __restrict__ A) {
    int r = blockIdx.x * blockDim.x + threadIdx.x;
    if (r >= ncenter * ns) return;
    int c = r / ns, p = gidx[r];
    _Float16* row = A + (size_t)r * kpad;
    row[0] = (_Float16)(pxy[2 * p] - cxy[2 * c]);
    row[1] = (_Float16)(pxy[2 * p + 1] - cxy[2 * c + 1]);
    row[2] = (_Float16)0.0f;
    const _Float16* f = feat + (size_t)p * fdim;
    for (int k = 0; k < fdim; ++k) row[3 + k] = f[k];
    for (int k = 3 + fdim; k < kpad; ++k) row[k] = (_Float16)0.0f;
}

// Build SA3 input rows: [x, y, 0, feat(fdim), zeros] padded to kpad.
__global__ void build_a3_kernel(const float* __restrict__ cxy, const _Float16* __restrict__ feat,
                                int fdim, int n, int kpad, _Float16* __restrict__ A) {
    int r = blockIdx.x * blockDim.x + threadIdx.x;
    if (r >= n) return;
    _Float16* row = A + (size_t)r * kpad;
    row[0] = (_Float16)cxy[2 * r];
    row[1] = (_Float16)cxy[2 * r + 1];
    row[2] = (_Float16)0.0f;
    const _Float16* f = feat + (size_t)r * fdim;
    for (int k = 0; k < fdim; ++k) row[3 + k] = f[k];
    for (int k = 3 + fdim; k < kpad; ++k) row[k] = (_Float16)0.0f;
}

// ---------------------------------------------------------------------------
// WMMA GEMM + folded-BN affine + ReLU.
// A: M x kpad row-major f16.  Wcm: N x kpad row-major f16 (i.e. W^T, padded).
// One wave per 16x16 output tile; K-loop in steps of 32 via
// v_wmma_f32_16x16x32_f16.
//
// Fragment layouts per CDNA5 ISA (wave32):
//  A 16x32 f16: lane l holds row (l&15); dword d holds K pair
//      {0,2,4,6}[d<4] / {16,18,20,22}[d>=4]  +  8*(l>=16)
//  B 32x16 f16: same K order as A (table order);
//  lane l holds col (l&15); dword d holds the same K pair as A's dword d.
// Both map to 2x global_load_b128 per fragment with our buffers.
// ---------------------------------------------------------------------------
__global__ void __launch_bounds__(256)
gemm_wmma_kernel(const _Float16* __restrict__ A, const _Float16* __restrict__ Wcm,
                 const float* __restrict__ scale, const float* __restrict__ shift,
                 _Float16* __restrict__ Of16, float* __restrict__ Of32,
                 int M, int N, int kpad) {
    int lane = threadIdx.x & 31;
    int wave = (blockIdx.x * blockDim.x + threadIdx.x) >> 5;
    int ntn = N >> 4;
    int ntiles = (M >> 4) * ntn;
    if (wave >= ntiles) return;
    int tm = wave / ntn, tn = wave % ntn;
    int row = tm * 16 + (lane & 15);
    int col = tn * 16 + (lane & 15);
    int half = lane >> 4;             // 0: lanes 0-15, 1: lanes 16-31
    int kh2A = half << 2;             // A dword offset: 0 or 4
    int kh2B = half << 3;             // unused; B loads use offA (same as A)
    const unsigned* Ap = (const unsigned*)(A + (size_t)row * kpad);
    const unsigned* Bp = (const unsigned*)(Wcm + (size_t)col * kpad);
    v8f acc = {};
    for (int k0 = 0; k0 < kpad; k0 += 32) {
        union { v16h v; unsigned u[8]; } a, b;
        int kb = k0 >> 1;
#pragma unroll
        for (int i = 0; i < 8; ++i) {
            int offA = ((i < 4) ? i : i + 4) + kh2A;  // {0..3}+kh2A, {8..11}+kh2A
            a.u[i] = Ap[kb + offA];
            b.u[i] = Bp[kb + offA];                   // same K order as A
        }
        acc = __builtin_amdgcn_wmma_f32_16x16x32_f16(false, a.v, false, b.v,
                                                     (short)0, acc, false, false);
    }
    float s = scale[col], t = shift[col];
    int rbase = tm * 16 + (half << 3);
#pragma unroll
    for (int r = 0; r < 8; ++r) {
        float y = fmaxf(acc[r] * s + t, 0.0f);
        size_t o = (size_t)(rbase + r) * N + col;
        if (Of16) Of16[o] = (_Float16)y;
        if (Of32) Of32[o] = y;
    }
}

// Max over groups of G consecutive rows of an (ngroups*G) x C f32 matrix.
__global__ void maxpool_kernel(const float* __restrict__ in, int C, int G, int ngroups,
                               _Float16* __restrict__ outf16, float* __restrict__ outf32) {
    int t = blockIdx.x * blockDim.x + threadIdx.x;
    if (t >= ngroups * C) return;
    int g = t / C, c = t % C;
    const float* p = in + (size_t)g * G * C + c;
    float m = p[0];
    for (int j = 1; j < G; ++j) m = fmaxf(m, p[(size_t)j * C]);
    if (outf16) outf16[t] = (_Float16)m;
    if (outf32) outf32[t] = m;
}

// FC head: 1024 -> 512 -> 256 -> 7 with BN+ReLU on fc1/fc2, then log_softmax.
__global__ void head_kernel(const float* __restrict__ g3,
                            const float* W1, const float* b1, const float* g1,
                            const float* bb1, const float* m1, const float* v1,
                            const float* W2, const float* b2, const float* g2,
                            const float* bb2, const float* m2, const float* v2,
                            const float* W3, const float* b3,
                            float* __restrict__ logits) {
    __shared__ float h0[1024];
    __shared__ float h1[512];
    __shared__ float h2[256];
    __shared__ float l[8];
    int t = threadIdx.x;
    h0[t] = g3[t];
    __syncthreads();
    if (t < 512) {
        float acc = 0.0f;
        for (int k = 0; k < 1024; ++k) acc += h0[k] * W1[k * 512 + t];
        float s = g1[t] * rsqrtf(v1[t] + EPSV);
        h1[t] = fmaxf((acc + b1[t] - m1[t]) * s + bb1[t], 0.0f);
    }
    __syncthreads();
    if (t < 256) {
        float acc = 0.0f;
        for (int k = 0; k < 512; ++k) acc += h1[k] * W2[k * 256 + t];
        float s = g2[t] * rsqrtf(v2[t] + EPSV);
        h2[t] = fmaxf((acc + b2[t] - m2[t]) * s + bb2[t], 0.0f);
    }
    __syncthreads();
    if (t < 7) {
        float acc = b3[t];
        for (int k = 0; k < 256; ++k) acc += h2[k] * W3[k * 7 + t];
        l[t] = acc;
    }
    __syncthreads();
    if (t == 0) {
        float mx = l[0];
        for (int c = 1; c < 7; ++c) mx = fmaxf(mx, l[c]);
        float se = 0.0f;
        for (int c = 0; c < 7; ++c) se += expf(l[c] - mx);
        float lse = logf(se);
        for (int c = 0; c < 7; ++c) logits[c] = l[c] - mx - lse;
    }
}

__global__ void broadcast_kernel(const float* __restrict__ logits, float* __restrict__ out, int n) {
    int i = blockIdx.x * blockDim.x + threadIdx.x;
    if (i < 7 * n) out[i] = logits[i / n];
}

// ---------------------------------------------------------------------------
extern "C" void kernel_launch(void* const* d_in, const int* in_sizes, int n_in,
                              void* d_out, int out_size, void* d_ws, size_t ws_size,
                              hipStream_t stream) {
    (void)in_sizes; (void)n_in; (void)out_size; (void)ws_size;
    const float* xy0 = (const float*)d_in[0];  // 65536 x 2

    struct Layer { int base, cin, cout, kpad; };
    const Layer L[9] = {
        {1, 3, 64, 32},    {7, 64, 64, 64},    {13, 64, 128, 64},
        {19, 131, 128, 160}, {25, 128, 128, 128}, {31, 128, 256, 128},
        {37, 259, 256, 288}, {43, 256, 512, 256}, {49, 512, 1024, 512}
    };

    char* p = (char*)d_ws;
    auto carve = [&](size_t bytes) {
        void* r = (void*)p;
        p += (bytes + 255) & ~(size_t)255;
        return r;
    };
    _Float16* wcm[9]; float* sc[9]; float* sh[9];
    for (int i = 0; i < 9; ++i) {
        wcm[i] = (_Float16*)carve((size_t)L[i].cout * L[i].kpad * 2);
        sc[i]  = (float*)carve((size_t)L[i].cout * 4);
        sh[i]  = (float*)carve((size_t)L[i].cout * 4);
    }
    int*       fps1 = (int*)carve(512 * 4);
    int*       fps2 = (int*)carve(128 * 4);
    float*     nx1  = (float*)carve(512 * 2 * 4);
    float*     nx2  = (float*)carve(128 * 2 * 4);
    int*       gi1  = (int*)carve(512 * 32 * 4);
    int*       gi2  = (int*)carve(128 * 64 * 4);
    _Float16*  A1   = (_Float16*)carve((size_t)16384 * 32 * 2);
    _Float16*  a1a  = (_Float16*)carve((size_t)16384 * 64 * 2);
    _Float16*  a1b  = (_Float16*)carve((size_t)16384 * 64 * 2);
    float*     a1c  = (float*)carve((size_t)16384 * 128 * 4);
    _Float16*  p1   = (_Float16*)carve((size_t)512 * 128 * 2);
    _Float16*  A2   = (_Float16*)carve((size_t)8192 * 160 * 2);
    _Float16*  a2a  = (_Float16*)carve((size_t)8192 * 128 * 2);
    _Float16*  a2b  = (_Float16*)carve((size_t)8192 * 128 * 2);
    float*     a2c  = (float*)carve((size_t)8192 * 256 * 4);
    _Float16*  p2   = (_Float16*)carve((size_t)128 * 256 * 2);
    _Float16*  A3   = (_Float16*)carve((size_t)128 * 288 * 2);
    _Float16*  a3a  = (_Float16*)carve((size_t)128 * 256 * 2);
    _Float16*  a3b  = (_Float16*)carve((size_t)128 * 512 * 2);
    float*     a3c  = (float*)carve((size_t)128 * 1024 * 4);
    float*     g3   = (float*)carve(1024 * 4);
    float*     lgts = (float*)carve(8 * 4);

    auto cdiv = [](int a, int b) { return (a + b - 1) / b; };

    // Fold BN + transpose/pad weights (f16 column-major).
    for (int i = 0; i < 9; ++i) {
        prep_layer_kernel<<<cdiv(L[i].cout * L[i].kpad, 256), 256, 0, stream>>>(
            (const float*)d_in[L[i].base + 0], (const float*)d_in[L[i].base + 1],
            (const float*)d_in[L[i].base + 2], (const float*)d_in[L[i].base + 3],
            (const float*)d_in[L[i].base + 4], (const float*)d_in[L[i].base + 5],
            L[i].cin, L[i].cout, L[i].kpad, wcm[i], sc[i], sh[i]);
    }

    auto gemm = [&](const _Float16* A, int li, _Float16* of16, float* of32, int M, int N) {
        int tiles = (M / 16) * (N / 16);
        gemm_wmma_kernel<<<cdiv(tiles, 8), 256, 0, stream>>>(
            A, wcm[li], sc[li], sh[li], of16, of32, M, N, L[li].kpad);
    };

    // ---- SA1 ----
    fps_kernel<<<1, 1024, 65536 * 4, stream>>>(xy0, 65536, 512, fps1);  // 256KB LDS dist
    gather_xy_kernel<<<cdiv(512, 256), 256, 0, stream>>>(xy0, fps1, 512, nx1);
    group_kernel<<<cdiv(512, 256), 256, 0, stream>>>(nx1, 512, xy0, 65536, 0.04f, 32, gi1);
    build_a1_kernel<<<cdiv(16384, 256), 256, 0, stream>>>(xy0, nx1, gi1, 512, 32, A1);
    gemm(A1, 0, a1a, nullptr, 16384, 64);
    gemm(a1a, 1, a1b, nullptr, 16384, 64);
    gemm(a1b, 2, nullptr, a1c, 16384, 128);
    maxpool_kernel<<<cdiv(512 * 128, 256), 256, 0, stream>>>(a1c, 128, 32, 512, p1, nullptr);

    // ---- SA2 ----
    fps_kernel<<<1, 1024, 512 * 4, stream>>>(nx1, 512, 128, fps2);
    gather_xy_kernel<<<1, 128, 0, stream>>>(nx1, fps2, 128, nx2);
    group_kernel<<<1, 128, 0, stream>>>(nx2, 128, nx1, 512, 0.16f, 64, gi2);
    build_a2_kernel<<<cdiv(8192, 256), 256, 0, stream>>>(nx1, nx2, gi2, p1, 128, 128, 64, 160, A2);
    gemm(A2, 3, a2a, nullptr, 8192, 128);
    gemm(a2a, 4, a2b, nullptr, 8192, 128);
    gemm(a2b, 5, nullptr, a2c, 8192, 256);
    maxpool_kernel<<<cdiv(128 * 256, 256), 256, 0, stream>>>(a2c, 256, 64, 128, p2, nullptr);

    // ---- SA3 (group_all) ----
    build_a3_kernel<<<1, 128, 0, stream>>>(nx2, p2, 256, 128, 288, A3);
    gemm(A3, 6, a3a, nullptr, 128, 256);
    gemm(a3a, 7, a3b, nullptr, 128, 512);
    gemm(a3b, 8, nullptr, a3c, 128, 1024);
    maxpool_kernel<<<cdiv(1024, 256), 256, 0, stream>>>(a3c, 1024, 128, 1, nullptr, g3);

    // ---- FC head + log_softmax ----
    head_kernel<<<1, 1024, 0, stream>>>(g3,
        (const float*)d_in[55], (const float*)d_in[56], (const float*)d_in[57],
        (const float*)d_in[58], (const float*)d_in[59], (const float*)d_in[60],
        (const float*)d_in[61], (const float*)d_in[62], (const float*)d_in[63],
        (const float*)d_in[64], (const float*)d_in[65], (const float*)d_in[66],
        (const float*)d_in[67], (const float*)d_in[68],
        lgts);

    // ---- broadcast logits -> (7, 65536) output (mu rows 0..3, lam rows 4..6) ----
    broadcast_kernel<<<cdiv(7 * 65536, 256), 256, 0, stream>>>(lgts, (float*)d_out, 65536);
}
